// IndRNN_14181982012201
// MI455X (gfx1250) — matrix operand order repaired and checked by reading, not compiled, not run on hardware
//
#include <hip/hip_runtime.h>
#include <hip/hip_bf16.h>

typedef __bf16 bf16;
typedef __attribute__((ext_vector_type(4)))  __bf16 v4bf;
typedef __attribute__((ext_vector_type(8)))  __bf16 v8bf;
typedef __attribute__((ext_vector_type(16))) __bf16 v16bf;
typedef __attribute__((ext_vector_type(8)))  float  v8f;

#define SEQ  2048
#define BATCH 32
#define XSIZE 256
#define HID  512
#define BH   (BATCH * HID)           // 16384 recurrence lanes
#define MROWS (SEQ * BATCH)          // 65536 GEMM rows

// ---------------------------------------------------------------------------
// f32 -> bf16 conversion, 4 elements per thread (16B load / 8B store)
// ---------------------------------------------------------------------------
__global__ void __launch_bounds__(256)
cvt_f32_to_bf16(const float* __restrict__ in, bf16* __restrict__ out, int n4) {
    int i = blockIdx.x * blockDim.x + threadIdx.x;
    if (i >= n4) return;
    const float4 v = *(const float4*)(in + (size_t)i * 4);
    v4bf o;
    o.x = (bf16)v.x; o.y = (bf16)v.y; o.z = (bf16)v.z; o.w = (bf16)v.w;
    *(v4bf*)(out + (size_t)i * 4) = o;
}

// ---------------------------------------------------------------------------
// bf16 WMMA GEMM:  C[m][n] = sum_k A[m][k] * W[n][k] + bias[n]
// A: MxK row-major bf16, W: NxK row-major bf16 (both K-contiguous -> NT GEMM)
// One wave computes a 64x64 output tile: 4x4 grid of 16x16 WMMA accumulators.
// ---------------------------------------------------------------------------
__global__ void __launch_bounds__(256)
gemm_bf16_nt(const bf16* __restrict__ A, const bf16* __restrict__ W,
             const float* __restrict__ bias, float* __restrict__ C,
             int M, int N, int K) {
    const int wave = blockIdx.x * (blockDim.x >> 5) + (threadIdx.x >> 5);
    const int lane = threadIdx.x & 31;
    const int l15  = lane & 15;
    const int lhi  = lane >> 4;          // 0 for lanes 0-15, 1 for lanes 16-31

    const int tilesN = N >> 6;
    const int m0 = (wave / tilesN) << 6;
    const int n0 = (wave % tilesN) << 6;
    if (m0 >= M) return;

    v8f acc[4][4] = {};

    for (int k = 0; k < K; k += 32) {
        v16bf af[4], bfr[4];
#pragma unroll
        for (int i = 0; i < 4; ++i) {
            // A fragment (16x32, MxK): lane holds row m = l15, two 16B chunks
            // at K offsets {lhi*16, lhi*16+8}  (contiguous, matches B)
            const bf16* ap = A + (size_t)(m0 + i * 16 + l15) * K + k + lhi * 16;
            v8bf alo = *(const v8bf*)ap;
            v8bf ahi = *(const v8bf*)(ap + 8);
            af[i] = __builtin_shufflevector(alo, ahi,
                     0, 1, 2, 3, 4, 5, 6, 7, 8, 9, 10, 11, 12, 13, 14, 15);
            // B fragment (32x16, KxN): lane holds column n = l15, one 32B
            // contiguous chunk at K offset lhi*16 (row n of W = column n of B)
            const bf16* bp = W + (size_t)(n0 + i * 16 + l15) * K + k + lhi * 16;
            bfr[i] = *(const v16bf*)bp;
        }
#pragma unroll
        for (int tm = 0; tm < 4; ++tm)
#pragma unroll
            for (int tn = 0; tn < 4; ++tn)
                acc[tm][tn] = __builtin_amdgcn_wmma_f32_16x16x32_bf16(
                    false, af[tm], false, bfr[tn],
                    (short)0, acc[tm][tn], false, false);
    }

    // Epilogue: C/D layout (ISA 7.12.2): VGPR r, lane L -> M = r + (L>=16)*8,
    // N = L&15. Add bias[n] before storing.
#pragma unroll
    for (int tn = 0; tn < 4; ++tn) {
        const int col = n0 + tn * 16 + l15;
        const float bv = bias[col];
#pragma unroll
        for (int tm = 0; tm < 4; ++tm) {
            const int rowBase = m0 + tm * 16 + lhi * 8;
#pragma unroll
            for (int r = 0; r < 8; ++r)
                C[(size_t)(rowBase + r) * N + col] = acc[tm][tn][r] + bv;
        }
    }
}

// ---------------------------------------------------------------------------
// IndRNN scan, layer 0: h = relu6(lin + rec*h); emit y as bf16 (feeds GEMM1),
// final h (f32) to hlast.
// ---------------------------------------------------------------------------
__global__ void __launch_bounds__(64)
indrnn_scan_bf16out(const float* __restrict__ lin, const float* __restrict__ rec,
                    bf16* __restrict__ yb, float* __restrict__ hlast) {
    const int idx = blockIdx.x * blockDim.x + threadIdx.x;  // 0..BH-1
    if (idx >= BH) return;
    const float r = rec[idx & (HID - 1)];
    float h = 0.0f;
    const float* p = lin + idx;
    bf16* q = yb + idx;
#pragma unroll 8
    for (int s = 0; s < SEQ; ++s) {
        float l = p[(size_t)s * BH];
        h = __builtin_fmaf(r, h, l);
        h = h < 0.0f ? 0.0f : (h > 6.0f ? 6.0f : h);
        q[(size_t)s * BH] = (bf16)h;
    }
    hlast[idx] = h;
}

// ---------------------------------------------------------------------------
// IndRNN scan, layer 1: in-place f32 (lin1 lives in d_out; overwrite with y1)
// ---------------------------------------------------------------------------
__global__ void __launch_bounds__(64)
indrnn_scan_inplace(float* __restrict__ buf, const float* __restrict__ rec,
                    float* __restrict__ hlast) {
    const int idx = blockIdx.x * blockDim.x + threadIdx.x;
    if (idx >= BH) return;
    const float r = rec[idx & (HID - 1)];
    float h = 0.0f;
    float* p = buf + idx;
#pragma unroll 8
    for (int s = 0; s < SEQ; ++s) {
        float l = p[(size_t)s * BH];
        h = __builtin_fmaf(r, h, l);
        h = h < 0.0f ? 0.0f : (h > 6.0f ? 6.0f : h);
        p[(size_t)s * BH] = h;
    }
    hlast[idx] = h;
}

// ---------------------------------------------------------------------------
extern "C" void kernel_launch(void* const* d_in, const int* in_sizes, int n_in,
                              void* d_out, int out_size, void* d_ws, size_t ws_size,
                              hipStream_t stream) {
    const float* x    = (const float*)d_in[0];   // (SEQ, BATCH, XSIZE)
    const float* W0   = (const float*)d_in[1];   // (HID, XSIZE)
    const float* b0   = (const float*)d_in[2];   // (HID,)
    const float* rec0 = (const float*)d_in[3];   // (HID,)
    const float* W1   = (const float*)d_in[4];   // (HID, HID)
    const float* b1   = (const float*)d_in[5];   // (HID,)
    const float* rec1 = (const float*)d_in[6];   // (HID,)

    float* out   = (float*)d_out;                         // (SEQ, BATCH, HID)
    float* hlast = out + (size_t)SEQ * BH;                // (2, BATCH, HID)

    // workspace layout (all offsets 256B-aligned)
    char* ws = (char*)d_ws;
    bf16* xb   = (bf16*)(ws);                                  // 33,554,432 B
    bf16* w0b  = (bf16*)(ws + 33554432);                       //    262,144 B
    bf16* w1b  = (bf16*)(ws + 33816576);                       //    524,288 B
    float* lin0 = (float*)(ws + 34340864);                     // 134,217,728 B
    bf16* y0b  = (bf16*)(ws + 34340864 + 134217728);           // 67,108,864 B

    // 1) cast inputs to bf16
    {
        int n4 = (MROWS * XSIZE) / 4;                          // 4,194,304
        cvt_f32_to_bf16<<<n4 / 256, 256, 0, stream>>>(x, xb, n4);
        n4 = (HID * XSIZE) / 4;                                // 32,768
        cvt_f32_to_bf16<<<n4 / 256, 256, 0, stream>>>(W0, w0b, n4);
        n4 = (HID * HID) / 4;                                  // 65,536
        cvt_f32_to_bf16<<<n4 / 256, 256, 0, stream>>>(W1, w1b, n4);
    }

    // 2) GEMM0: lin0 = x @ W0^T + b0   (65536 x 256) x (256 x 512)
    {
        const int waves = (MROWS / 64) * (HID / 64);           // 8192
        gemm_bf16_nt<<<waves / 8, 256, 0, stream>>>(xb, w0b, b0, lin0,
                                                    MROWS, HID, XSIZE);
    }

    // 3) scan layer 0 -> y0 (bf16) + h0
    indrnn_scan_bf16out<<<BH / 64, 64, 0, stream>>>(lin0, rec0, y0b, hlast);

    // 4) GEMM1: lin1 = y0 @ W1^T + b1 -> written directly into d_out
    {
        const int waves = (MROWS / 64) * (HID / 64);           // 8192
        gemm_bf16_nt<<<waves / 8, 256, 0, stream>>>(y0b, w1b, b1, out,
                                                    MROWS, HID, HID);
    }

    // 5) scan layer 1 in-place in d_out -> y1 + h1
    indrnn_scan_inplace<<<BH / 64, 64, 0, stream>>>(out, rec1, hlast + BH);
}
